// Jitter_model_2894807957882
// MI455X (gfx1250) — matrix operand order repaired and checked by reading, not compiled, not run on hardware
//
#include <hip/hip_runtime.h>
#include <hip/hip_bf16.h>

// ---------------------------------------------------------------------------
// RouteNet-style jitter GNN for MI455X (gfx1250).
// Heavy matmuls ([Mx128]@[128x384] GRU gates, [Mx128]@[128x128] embeddings)
// use bf16 WMMA (v_wmma_f32_16x16x32_bf16) with f32 accumulate.
// B operand is staged into LDS once per 8-wave workgroup via the CDNA5
// async global->LDS path (global_load_async_to_lds_b128 + s_wait_asynccnt),
// A fragments stay resident in VGPRs across all N blocks.
// ---------------------------------------------------------------------------

typedef __bf16 bf16_t;
typedef __attribute__((ext_vector_type(8)))  __bf16 v8bf;
typedef __attribute__((ext_vector_type(16))) __bf16 v16bf;
typedef __attribute__((ext_vector_type(8)))  float  v8f;

#define PN    8000
#define LN    8
#define NLN   2000
#define NNN   500
#define KLINK 32
#define KNODE 64
#define KLN   16
#define DD    128
#define NITER 12

__device__ __forceinline__ bf16_t f2bf(float f) {
  unsigned int u = __float_as_uint(f);
  unsigned int r = (u + 0x7FFFu + ((u >> 16) & 1u)) >> 16;   // RNE
  unsigned short s = (unsigned short)r;
  return __builtin_bit_cast(bf16_t, s);
}
__device__ __forceinline__ float bf2f(bf16_t b) {
  unsigned short s = __builtin_bit_cast(unsigned short, b);
  return __uint_as_float(((unsigned int)s) << 16);
}

// ---------------------------------------------------------------------------
// WMMA fragment loaders (CDNA5 ISA 7.12.2 layouts, wave32).
// A (16x32 bf16, row-major MxK source): lane m=l&15, half=l>>4 holds
//   K = {8h..8h+7} U {16+8h..16+8h+7}  -> two contiguous 16B loads.
// B (32x16 bf16) from transposed Bt[N][K]: lane n=l&15, half=l>>4 holds
//   K = {8h..8h+7} U {16+8h..16+8h+7} -> two contiguous 16B loads.
// D/C f32: VGPR i <-> row m0 + i + 8*half, col = n0 + (l&15).
// ---------------------------------------------------------------------------
__device__ __forceinline__ v16bf load_frag_a(const bf16_t* __restrict__ rowp,
                                             int kb, int half) {
  v8bf lo = *(const v8bf*)(rowp + kb + 8 * half);
  v8bf hi = *(const v8bf*)(rowp + kb + 16 + 8 * half);
  v16bf a;
#pragma unroll
  for (int i = 0; i < 8; ++i) { a[i] = lo[i]; a[i + 8] = hi[i]; }
  return a;
}
__device__ __forceinline__ v16bf load_frag_b_lds(const bf16_t* colp, int kb, int half) {
  v8bf lo = *(const v8bf*)(colp + kb + 8 * half);
  v8bf hi = *(const v8bf*)(colp + kb + 16 + 8 * half);
  v16bf b;
#pragma unroll
  for (int i = 0; i < 8; ++i) { b[i] = lo[i]; b[i + 8] = hi[i]; }
  return b;
}
__device__ __forceinline__ void store_tile(v8f acc, int m0, int half, int col,
                                           const float* __restrict__ bias,
                                           float* __restrict__ C,
                                           bf16_t* __restrict__ Cbf,
                                           int M, int N, int relu) {
  float bv = bias ? bias[col] : 0.0f;
#pragma unroll
  for (int i = 0; i < 8; ++i) {
    int r = m0 + i + 8 * half;
    if (r < M) {
      float v = acc[i] + bv;
      if (relu) v = v > 0.0f ? v : 0.0f;
      C[(size_t)r * N + col] = v;
      if (Cbf) Cbf[(size_t)r * N + col] = f2bf(v);
    }
  }
}

// C[MxNT] = relu?( A[Mx128](bf16) @ Bt[NTx128](bf16)^T + bias ), opt bf16 copy.
// Block = 256 threads (8 waves), one block per 128-row band; B panel staged
// once in LDS by async global->LDS copies and shared by all 8 waves.
template <int NT>
__global__ __launch_bounds__(256) void gemm_bf16_wmma_lds(
    const bf16_t* __restrict__ A, const bf16_t* __restrict__ Bt,
    const float* __restrict__ bias, float* __restrict__ C,
    bf16_t* __restrict__ Cbf, int M, int relu) {
  constexpr int K = 128;
  __shared__ bf16_t sB[NT * K];               // 384x128: 96 KB, 128x128: 32 KB

  const int tid  = threadIdx.x;
  const int wave = tid >> 5;
  const int lane = tid & 31;
  const int half = lane >> 4;
  const int lm   = lane & 15;

  // ---- async stage Bt panel into LDS (ASYNCcnt path) ----------------------
  {
    const unsigned ldsbase = (unsigned)(uintptr_t)(&sB[0]);
    constexpr int chunks = (NT * K) / 8;      // 16-byte chunks; multiple of 256
#pragma unroll 1
    for (int c = tid; c < chunks; c += 256) {
      const bf16_t* gp = Bt + (size_t)c * 8;
      unsigned laddr = ldsbase + (unsigned)c * 16u;
      asm volatile("global_load_async_to_lds_b128 %0, %1, off"
                   :: "v"(laddr), "v"(gp) : "memory");
    }
    asm volatile("s_wait_asynccnt 0x0" ::: "memory");
  }
  __syncthreads();

  // ---- A fragments for this wave's 16-row band, resident in VGPRs ---------
  const int m0 = blockIdx.x * 128 + wave * 16;
  int row = m0 + lm;
  if (row >= M) row = M - 1;                  // clamp (tail band)
  const bf16_t* arow = A + (size_t)row * K;
  v16bf afrag[4];
#pragma unroll
  for (int ki = 0; ki < 4; ++ki) afrag[ki] = load_frag_a(arow, ki * 32, half);

  // ---- N-block loop: B from LDS, 4 accumulators per 64-wide block ---------
#pragma unroll 1
  for (int nb = 0; nb < NT; nb += 64) {
    const bf16_t* b0p = sB + (size_t)(nb +  0 + lm) * K;
    const bf16_t* b1p = sB + (size_t)(nb + 16 + lm) * K;
    const bf16_t* b2p = sB + (size_t)(nb + 32 + lm) * K;
    const bf16_t* b3p = sB + (size_t)(nb + 48 + lm) * K;
    v8f acc0 = {}, acc1 = {}, acc2 = {}, acc3 = {};
#pragma unroll
    for (int ki = 0; ki < 4; ++ki) {
      const int kb = ki * 32;
      v16bf b0 = load_frag_b_lds(b0p, kb, half);
      v16bf b1 = load_frag_b_lds(b1p, kb, half);
      v16bf b2 = load_frag_b_lds(b2p, kb, half);
      v16bf b3 = load_frag_b_lds(b3p, kb, half);
      acc0 = __builtin_amdgcn_wmma_f32_16x16x32_bf16(false, afrag[ki], false, b0, (short)0, acc0, false, false);
      acc1 = __builtin_amdgcn_wmma_f32_16x16x32_bf16(false, afrag[ki], false, b1, (short)0, acc1, false, false);
      acc2 = __builtin_amdgcn_wmma_f32_16x16x32_bf16(false, afrag[ki], false, b2, (short)0, acc2, false, false);
      acc3 = __builtin_amdgcn_wmma_f32_16x16x32_bf16(false, afrag[ki], false, b3, (short)0, acc3, false, false);
    }
    store_tile(acc0, m0, half, nb +  0 + lm, bias, C, Cbf, M, NT, relu);
    store_tile(acc1, m0, half, nb + 16 + lm, bias, C, Cbf, M, NT, relu);
    store_tile(acc2, m0, half, nb + 32 + lm, bias, C, Cbf, M, NT, relu);
    store_tile(acc3, m0, half, nb + 48 + lm, bias, C, Cbf, M, NT, relu);
  }
}

// --------------------------- helper kernels --------------------------------
__global__ void k_transpose_bf(const float* __restrict__ W, bf16_t* __restrict__ Wt,
                               int K, int N) {
  int i = blockIdx.x * blockDim.x + threadIdx.x;
  if (i >= K * N) return;
  int k = i / N, n = i % N;
  Wt[(size_t)n * K + k] = f2bf(W[i]);
}

__global__ void k_copy_f32(const float* __restrict__ s, float* __restrict__ d, int n) {
  int i = blockIdx.x * blockDim.x + threadIdx.x;
  if (i < n) d[i] = s[i];
}

// seq[p][0][d] = bf16(path_state[p][d])
__global__ void k_seq_slot0(const float* __restrict__ ps, bf16_t* __restrict__ seq) {
  int i = blockIdx.x * blockDim.x + threadIdx.x;
  if (i >= PN * DD) return;
  int p = i / DD, d = i % DD;
  seq[((size_t)p * (LN + 1)) * DD + d] = f2bf(ps[i]);
}

// msg[p][d] = bf16( link_state[l2p[p][t]][d] + dev_state[n2p[p][t]][d] )
__global__ void k_gather_msg(const float* __restrict__ ls, const float* __restrict__ ds,
                             const int* __restrict__ l2p, const int* __restrict__ n2p,
                             bf16_t* __restrict__ out, int t) {
  int i = blockIdx.x * blockDim.x + threadIdx.x;
  if (i >= PN * DD) return;
  int p = i / DD, d = i % DD;
  int l = l2p[p * LN + t];
  int n = n2p[p * LN + t];
  out[i] = f2bf(ls[(size_t)l * DD + d] + ds[(size_t)n * DD + d]);
}

// out[r][d] = bf16( sum_j seq[idx2[r][j][0]][idx2[r][j][1]][d] )
__global__ void k_gather_sum(const bf16_t* __restrict__ seq, const int* __restrict__ idx2,
                             bf16_t* __restrict__ out, int rows, int Kg) {
  int i = blockIdx.x * blockDim.x + threadIdx.x;
  if (i >= rows * DD) return;
  int r = i / DD, d = i % DD;
  float s = 0.0f;
  for (int j = 0; j < Kg; ++j) {
    int p  = idx2[(r * Kg + j) * 2 + 0];
    int sl = idx2[(r * Kg + j) * 2 + 1];
    s += bf2f(seq[((size_t)p * (LN + 1) + sl) * DD + d]);
  }
  out[i] = f2bf(s);
}

// GRU gate math. mx/mh are [M x 3*DD]. Updates h in place, bf16 shadow,
// optional write of the new state into sequence slot `slot`.
__global__ void k_gru_pointwise(const float* __restrict__ mx, const float* __restrict__ mh,
                                float* __restrict__ h, bf16_t* __restrict__ hbf,
                                bf16_t* __restrict__ seq, int slot, int M) {
  int i = blockIdx.x * blockDim.x + threadIdx.x;
  if (i >= M * DD) return;
  int m = i / DD, d = i % DD;
  size_t b = (size_t)m * 3 * DD + d;
  float xz = mx[b], xr = mx[b + DD], xh = mx[b + 2 * DD];
  float hz = mh[b], hr = mh[b + DD], hh = mh[b + 2 * DD];
  float z = 1.0f / (1.0f + __expf(-(xz + hz)));
  float r = 1.0f / (1.0f + __expf(-(xr + hr)));
  float c = tanhf(xh + r * hh);
  float hn = z * h[i] + (1.0f - z) * c;
  h[i]   = hn;
  hbf[i] = f2bf(hn);
  if (seq) seq[((size_t)m * (LN + 1) + slot) * DD + d] = f2bf(hn);
}

// ----- feature / embedding layer-1 kernels (K is tiny: 2-3; scalar is fine) -----
__global__ void k_emb1_path(const float* __restrict__ tr, const float* __restrict__ pk,
                            const float* __restrict__ sz, const float* __restrict__ W,
                            const float* __restrict__ b, bf16_t* __restrict__ out) {
  int i = blockIdx.x * blockDim.x + threadIdx.x;
  if (i >= PN * DD) return;
  int p = i / DD, d = i % DD;
  float v = tr[p] * W[d] + pk[p] * W[DD + d] + sz[p] * W[2 * DD + d] + b[d];
  out[i] = f2bf(v > 0.0f ? v : 0.0f);
}
__global__ void k_link_load(const float* __restrict__ tr, const int* __restrict__ p2l,
                            const float* __restrict__ cap, float* __restrict__ load) {
  int l = blockIdx.x * blockDim.x + threadIdx.x;
  if (l >= NLN) return;
  float s = 0.0f;
  for (int j = 0; j < KLINK; ++j) s += tr[p2l[(l * KLINK + j) * 2 + 0]];
  load[l] = s / (cap[l] * 1.0e9f);
}
__global__ void k_emb1_link(const float* __restrict__ cap, const float* __restrict__ load,
                            const int* __restrict__ ldt, const float* __restrict__ W,
                            const float* __restrict__ b, bf16_t* __restrict__ out) {
  int i = blockIdx.x * blockDim.x + threadIdx.x;
  if (i >= NLN * DD) return;
  int l = i / DD, d = i % DD;
  float f2 = (ldt[l] == 0) ? 1.0f : 0.0f;
  float v = cap[l] * W[d] + load[l] * W[DD + d] + f2 * W[2 * DD + d] + b[d];
  out[i] = f2bf(v > 0.0f ? v : 0.0f);
}
__global__ void k_dev_mean(const float* __restrict__ ls, const int* __restrict__ l2n,
                           float* __restrict__ dm) {
  int n = blockIdx.x * blockDim.x + threadIdx.x;
  if (n >= NNN) return;
  float s = 0.0f;
  for (int j = 0; j < KLN; ++j) {
    const float* row = ls + (size_t)l2n[n * KLN + j] * DD;
    for (int d = 0; d < DD; ++d) s += row[d];
  }
  dm[n] = s / (float)DD;
}
__global__ void k_emb1_dev(const int* __restrict__ nodes, const float* __restrict__ dm,
                           const float* __restrict__ W, const float* __restrict__ b,
                           bf16_t* __restrict__ out) {
  int i = blockIdx.x * blockDim.x + threadIdx.x;
  if (i >= NNN * DD) return;
  int n = i / DD, d = i % DD;
  float f0 = (nodes[n] == 0) ? 1.0f : 0.0f;
  float v = f0 * W[d] + dm[n] * W[DD + d] + b[d];
  out[i] = f2bf(v > 0.0f ? v : 0.0f);
}

// ----- readout: per-path, 8 positions, MLP 3->64->32->32->16->1 (last linear) -----
__global__ void k_readout(const bf16_t* __restrict__ seq, const float* __restrict__ lstop,
                          const int* __restrict__ l2p, const float* __restrict__ cap,
                          const float* __restrict__ W1, const float* __restrict__ b1,
                          const float* __restrict__ W2, const float* __restrict__ b2,
                          const float* __restrict__ W3, const float* __restrict__ b3,
                          const float* __restrict__ W4, const float* __restrict__ b4,
                          const float* __restrict__ W5, const float* __restrict__ b5,
                          float* __restrict__ out) {
  int p = blockIdx.x * blockDim.x + threadIdx.x;
  if (p >= PN) return;
  float acc = 0.0f;
  for (int t = 0; t < LN; ++t) {
    int l = l2p[p * LN + t];
    float f0 = cap[l];
    float f1 = -3.4e38f;
    const bf16_t* sp = seq + ((size_t)p * (LN + 1) + t + 1) * DD;
    for (int d = 0; d < DD; ++d) { float v = bf2f(sp[d]); f1 = v > f1 ? v : f1; }
    float f2 = 0.0f;
    const float* lp = lstop + (size_t)l * DD;
    for (int d = 0; d < DD; ++d) f2 += lp[d];
    float h1[64], h2[32], h3[32], h4[16];
    for (int j = 0; j < 64; ++j) {
      float v = f0 * W1[j] + f1 * W1[64 + j] + f2 * W1[128 + j] + b1[j];
      h1[j] = v > 0.0f ? v : 0.0f;
    }
    for (int j = 0; j < 32; ++j) {
      float v = b2[j];
      for (int k = 0; k < 64; ++k) v += h1[k] * W2[k * 32 + j];
      h2[j] = v > 0.0f ? v : 0.0f;
    }
    for (int j = 0; j < 32; ++j) {
      float v = b3[j];
      for (int k = 0; k < 32; ++k) v += h2[k] * W3[k * 32 + j];
      h3[j] = v > 0.0f ? v : 0.0f;
    }
    for (int j = 0; j < 16; ++j) {
      float v = b4[j];
      for (int k = 0; k < 32; ++k) v += h3[k] * W4[k * 16 + j];
      h4[j] = v > 0.0f ? v : 0.0f;
    }
    float o = b5[0];
    for (int k = 0; k < 16; ++k) o += h4[k] * W5[k];
    acc += o;
  }
  out[p] = acc;
}

// ---------------------------------------------------------------------------
extern "C" void kernel_launch(void* const* d_in, const int* in_sizes, int n_in,
                              void* d_out, int out_size, void* d_ws, size_t ws_size,
                              hipStream_t stream) {
  (void)in_sizes; (void)n_in; (void)out_size; (void)ws_size;
  // Top-level dict in setup_inputs() insertion order; nested params pytree in
  // JAX flatten order (dict keys sorted).
  const float* flow_traffic = (const float*)d_in[0];
  const float* flow_packets = (const float*)d_in[1];
  const float* flow_psize   = (const float*)d_in[2];
  const float* link_cap     = (const float*)d_in[3];
  const float* dev_emb_W1   = (const float*)d_in[4];
  const float* dev_emb_b1   = (const float*)d_in[5];
  const float* dev_emb_W2   = (const float*)d_in[6];
  const float* dev_emb_b2   = (const float*)d_in[7];
  const float* dev_gru_bi   = (const float*)d_in[8];
  const float* dev_gru_br   = (const float*)d_in[9];
  const float* dev_gru_K    = (const float*)d_in[10];
  const float* dev_gru_R    = (const float*)d_in[11];
  const float* link_emb_W1  = (const float*)d_in[12];
  const float* link_emb_b1  = (const float*)d_in[13];
  const float* link_emb_W2  = (const float*)d_in[14];
  const float* link_emb_b2  = (const float*)d_in[15];
  const float* link_gru_bi  = (const float*)d_in[16];
  const float* link_gru_br  = (const float*)d_in[17];
  const float* link_gru_K   = (const float*)d_in[18];
  const float* link_gru_R   = (const float*)d_in[19];
  const float* path_emb_W1  = (const float*)d_in[20];
  const float* path_emb_b1  = (const float*)d_in[21];
  const float* path_emb_W2  = (const float*)d_in[22];
  const float* path_emb_b2  = (const float*)d_in[23];
  const float* path_gru_bi  = (const float*)d_in[24];
  const float* path_gru_br  = (const float*)d_in[25];
  const float* path_gru_K   = (const float*)d_in[26];
  const float* path_gru_R   = (const float*)d_in[27];
  const float* ro_W1 = (const float*)d_in[28]; const float* ro_b1 = (const float*)d_in[29];
  const float* ro_W2 = (const float*)d_in[30]; const float* ro_b2 = (const float*)d_in[31];
  const float* ro_W3 = (const float*)d_in[32]; const float* ro_b3 = (const float*)d_in[33];
  const float* ro_W4 = (const float*)d_in[34]; const float* ro_b4 = (const float*)d_in[35];
  const float* ro_W5 = (const float*)d_in[36]; const float* ro_b5 = (const float*)d_in[37];
  const int* link_to_path = (const int*)d_in[38];
  const int* path_to_link = (const int*)d_in[39];
  const int* nodes        = (const int*)d_in[40];
  const int* link_to_node = (const int*)d_in[41];
  const int* link_dev_t   = (const int*)d_in[42];
  // d_in[43] = node_to_link (unused by the math)
  const int* node_to_path = (const int*)d_in[44];
  const int* path_to_node = (const int*)d_in[45];
  float* out = (float*)d_out;

  // ---- workspace carve-up -------------------------------------------------
  char* w = (char*)d_ws;
  auto take = [&](size_t bytes) {
    void* p = (void*)w;
    w += (bytes + 255) & ~(size_t)255;
    return p;
  };
  bf16_t* seq    = (bf16_t*)take(sizeof(short) * (size_t)PN * (LN + 1) * DD);
  float*  mx     = (float*) take(sizeof(float) * (size_t)PN * 3 * DD);
  float*  mh     = (float*) take(sizeof(float) * (size_t)PN * 3 * DD);
  float*  pstate = (float*) take(sizeof(float) * (size_t)PN * DD);
  bf16_t* psbf   = (bf16_t*)take(sizeof(short) * (size_t)PN * DD);
  bf16_t* abf    = (bf16_t*)take(sizeof(short) * (size_t)PN * DD);   // gather / layer1 staging
  float*  lstate = (float*) take(sizeof(float) * (size_t)NLN * DD);
  bf16_t* lsbf   = (bf16_t*)take(sizeof(short) * (size_t)NLN * DD);
  float*  lstop  = (float*) take(sizeof(float) * (size_t)NLN * DD);
  float*  dstate = (float*) take(sizeof(float) * (size_t)NNN * DD);
  bf16_t* dsbf   = (bf16_t*)take(sizeof(short) * (size_t)NNN * DD);
  float*  loadv  = (float*) take(sizeof(float) * NLN);
  float*  dmean  = (float*) take(sizeof(float) * NNN);
  bf16_t* pKt    = (bf16_t*)take(sizeof(short) * 384 * 128);
  bf16_t* pRt    = (bf16_t*)take(sizeof(short) * 384 * 128);
  bf16_t* lKt    = (bf16_t*)take(sizeof(short) * 384 * 128);
  bf16_t* lRt    = (bf16_t*)take(sizeof(short) * 384 * 128);
  bf16_t* dKt    = (bf16_t*)take(sizeof(short) * 384 * 128);
  bf16_t* dRt    = (bf16_t*)take(sizeof(short) * 384 * 128);
  bf16_t* pE2t   = (bf16_t*)take(sizeof(short) * 128 * 128);
  bf16_t* lE2t   = (bf16_t*)take(sizeof(short) * 128 * 128);
  bf16_t* dE2t   = (bf16_t*)take(sizeof(short) * 128 * 128);

  auto g = [](int n, int b) { return dim3((unsigned)((n + b - 1) / b)); };
  const int TB = 256;
  const dim3 gP((PN  + 127) / 128);   // 63 bands
  const dim3 gL((NLN + 127) / 128);   // 16 bands
  const dim3 gD((NNN + 127) / 128);   // 4 bands

  // ---- weight convert + transpose (bf16, [N x K]) -------------------------
  k_transpose_bf<<<g(128 * 384, TB), TB, 0, stream>>>(path_gru_K, pKt, 128, 384);
  k_transpose_bf<<<g(128 * 384, TB), TB, 0, stream>>>(path_gru_R, pRt, 128, 384);
  k_transpose_bf<<<g(128 * 384, TB), TB, 0, stream>>>(link_gru_K, lKt, 128, 384);
  k_transpose_bf<<<g(128 * 384, TB), TB, 0, stream>>>(link_gru_R, lRt, 128, 384);
  k_transpose_bf<<<g(128 * 384, TB), TB, 0, stream>>>(dev_gru_K,  dKt, 128, 384);
  k_transpose_bf<<<g(128 * 384, TB), TB, 0, stream>>>(dev_gru_R,  dRt, 128, 384);
  k_transpose_bf<<<g(128 * 128, TB), TB, 0, stream>>>(path_emb_W2, pE2t, 128, 128);
  k_transpose_bf<<<g(128 * 128, TB), TB, 0, stream>>>(link_emb_W2, lE2t, 128, 128);
  k_transpose_bf<<<g(128 * 128, TB), TB, 0, stream>>>(dev_emb_W2,  dE2t, 128, 128);

  // ---- embeddings ---------------------------------------------------------
  k_emb1_path<<<g(PN * DD, TB), TB, 0, stream>>>(flow_traffic, flow_packets, flow_psize,
                                                 path_emb_W1, path_emb_b1, abf);
  gemm_bf16_wmma_lds<128><<<gP, 256, 0, stream>>>(abf, pE2t, path_emb_b2,
                                                  pstate, psbf, PN, 1);
  k_link_load<<<g(NLN, TB), TB, 0, stream>>>(flow_traffic, path_to_link, link_cap, loadv);
  k_emb1_link<<<g(NLN * DD, TB), TB, 0, stream>>>(link_cap, loadv, link_dev_t,
                                                  link_emb_W1, link_emb_b1, abf);
  gemm_bf16_wmma_lds<128><<<gL, 256, 0, stream>>>(abf, lE2t, link_emb_b2,
                                                  lstate, lsbf, NLN, 1);
  k_dev_mean<<<g(NNN, TB), TB, 0, stream>>>(lstate, link_to_node, dmean);
  k_emb1_dev<<<g(NNN * DD, TB), TB, 0, stream>>>(nodes, dmean, dev_emb_W1, dev_emb_b1, abf);
  gemm_bf16_wmma_lds<128><<<gD, 256, 0, stream>>>(abf, dE2t, dev_emb_b2,
                                                  dstate, dsbf, NNN, 1);

  // ---- message passing iterations ----------------------------------------
  for (int it = 0; it < NITER; ++it) {
    // snapshot for readout link_g (state at top of iteration) + seq slot 0
    k_copy_f32<<<g(NLN * DD, TB), TB, 0, stream>>>(lstate, lstop, NLN * DD);
    k_seq_slot0<<<g(PN * DD, TB), TB, 0, stream>>>(pstate, seq);

    // path GRU scan over the 8 hops (sequential dependency)
    for (int t = 0; t < LN; ++t) {
      k_gather_msg<<<g(PN * DD, TB), TB, 0, stream>>>(lstate, dstate, link_to_path,
                                                      node_to_path, abf, t);
      gemm_bf16_wmma_lds<384><<<gP, 256, 0, stream>>>(abf, pKt, path_gru_bi,
                                                      mx, nullptr, PN, 0);
      gemm_bf16_wmma_lds<384><<<gP, 256, 0, stream>>>(psbf, pRt, path_gru_br,
                                                      mh, nullptr, PN, 0);
      k_gru_pointwise<<<g(PN * DD, TB), TB, 0, stream>>>(mx, mh, pstate, psbf,
                                                         seq, t + 1, PN);
    }

    // link GRU
    k_gather_sum<<<g(NLN * DD, TB), TB, 0, stream>>>(seq, path_to_link, abf, NLN, KLINK);
    gemm_bf16_wmma_lds<384><<<gL, 256, 0, stream>>>(abf, lKt, link_gru_bi,
                                                    mx, nullptr, NLN, 0);
    gemm_bf16_wmma_lds<384><<<gL, 256, 0, stream>>>(lsbf, lRt, link_gru_br,
                                                    mh, nullptr, NLN, 0);
    k_gru_pointwise<<<g(NLN * DD, TB), TB, 0, stream>>>(mx, mh, lstate, lsbf,
                                                        nullptr, 0, NLN);

    // device GRU
    k_gather_sum<<<g(NNN * DD, TB), TB, 0, stream>>>(seq, path_to_node, abf, NNN, KNODE);
    gemm_bf16_wmma_lds<384><<<gD, 256, 0, stream>>>(abf, dKt, dev_gru_bi,
                                                    mx, nullptr, NNN, 0);
    gemm_bf16_wmma_lds<384><<<gD, 256, 0, stream>>>(dsbf, dRt, dev_gru_br,
                                                    mh, nullptr, NNN, 0);
    k_gru_pointwise<<<g(NNN * DD, TB), TB, 0, stream>>>(mx, mh, dstate, dsbf,
                                                        nullptr, 0, NNN);
  }

  // ---- readout ------------------------------------------------------------
  k_readout<<<g(PN, 128), 128, 0, stream>>>(seq, lstop, link_to_path, link_cap,
                                            ro_W1, ro_b1, ro_W2, ro_b2, ro_W3, ro_b3,
                                            ro_W4, ro_b4, ro_W5, ro_b5, out);
}
